// DifferentiableSATSolver_81003083202771
// MI455X (gfx1250) — compile-verified, matrix operand order checked by reading, not run on hardware
//
#include <hip/hip_runtime.h>
#include <math.h>
#include <stdint.h>

typedef __attribute__((ext_vector_type(16))) _Float16 v16h;
typedef __attribute__((ext_vector_type(8)))  float    v8f;

#define BLOCK 256
#define CPT   16   // clauses per thread == f16 elems per lane in one 16x16x32 A-fragment

// ds_swizzle_b32 xor-lane exchange (group-of-32 mode: offset[14:10]=xor, [4:0]=and=0x1f)
#define SWZ_F32(x, imm) __int_as_float(__builtin_amdgcn_ds_swizzle(__float_as_int(x), (imm)))

// Kernel 1: assignments = sigmoid(logits); also (re)initialize min/count slots every
// call (harness poisons d_out once, never re-poisons -> we must init for determinism).
__global__ void sat_sigmoid_init(const float* __restrict__ logits,
                                 float* __restrict__ out,
                                 int n_vars, long long min_slot_idx) {
    long long i = (long long)blockIdx.x * blockDim.x + threadIdx.x;
    if (i < n_vars) {
        float x = logits[i];
        out[i] = 1.0f / (1.0f + expf(-x));
    }
    if (i == 0) {
        ((unsigned int*)out)[min_slot_idx] = 0x7F800000u; // +inf bits (uint order == float order for positives)
        out[min_slot_idx + 1] = 0.0f;                     // n_satisfied accumulator
    }
}

// Kernel 2: per-clause soft-OR (b128-vectorized streams), global min, WMMA count.
__global__ void sat_clause_kernel(const float* __restrict__ assign,
                                  const int*   __restrict__ vic,
                                  const int*   __restrict__ signs,
                                  float* __restrict__ sat_out,
                                  float* __restrict__ min_slot,
                                  float* __restrict__ cnt_slot,
                                  int n_clauses) {
    const int tid = threadIdx.x;
    const long long blockStart = (long long)blockIdx.x * (BLOCK * CPT);

    v16h aFrag;
    float localMin = __builtin_inff();

    // Uniform fast-path test: full block in range and 16B-alignable vector store.
    const bool fast = (blockStart + (long long)BLOCK * CPT <= (long long)n_clauses) &&
                      ((((uintptr_t)sat_out) & 15) == 0);

    if (fast) {
        // 4 groups of 4 contiguous clauses per thread: each group = 48B of vic,
        // 48B of signs (3x global_load_b128 each), 12 L2-resident gathers, 1 b128 store.
#pragma unroll
        for (int g = 0; g < 4; ++g) {
            const long long base = blockStart + ((long long)g * BLOCK + tid) * 4;
            const int4* v4 = reinterpret_cast<const int4*>(vic + 3 * base);   // 48B-aligned
            const int4* s4 = reinterpret_cast<const int4*>(signs + 3 * base);
            const int4 va = v4[0], vb = v4[1], vc = v4[2];
            const int4 sa = s4[0], sb = s4[1], sc = s4[2];
            int vv[12] = { va.x, va.y, va.z, va.w, vb.x, vb.y, vb.z, vb.w,
                           vc.x, vc.y, vc.z, vc.w };
            int ss[12] = { sa.x, sa.y, sa.z, sa.w, sb.x, sb.y, sb.z, sb.w,
                           sc.x, sc.y, sc.z, sc.w };
            float sat[4];
#pragma unroll
            for (int j = 0; j < 4; ++j) {
                float a0 = assign[vv[3 * j + 0]];
                float a1 = assign[vv[3 * j + 1]];
                float a2 = assign[vv[3 * j + 2]];
                float l0 = (ss[3 * j + 0] == 1) ? a0 : 1.0f - a0;
                float l1 = (ss[3 * j + 1] == 1) ? a1 : 1.0f - a1;
                float l2 = (ss[3 * j + 2] == 1) ? a2 : 1.0f - a2;
                sat[j] = fmaxf(fmaxf(l0, l1), l2);
                localMin = fminf(localMin, sat[j]);
                aFrag[g * 4 + j] = (sat[j] > 0.5f) ? (_Float16)1.0f : (_Float16)0.0f;
            }
            float4 o; o.x = sat[0]; o.y = sat[1]; o.z = sat[2]; o.w = sat[3];
            *reinterpret_cast<float4*>(sat_out + base) = o;   // global_store_b128
        }
    } else {
        // Tail / misaligned fallback: scalar per-clause (covers the single partial block).
#pragma unroll
        for (int t = 0; t < CPT; ++t) {
            long long c = blockStart + (long long)t * BLOCK + tid;
            _Float16 ind = (_Float16)0.0f;
            if (c < n_clauses) {
                long long b = 3ll * c;
                int v0 = vic[b], v1 = vic[b + 1], v2 = vic[b + 2];
                int s0 = signs[b], s1 = signs[b + 1], s2 = signs[b + 2];
                float a0 = assign[v0], a1 = assign[v1], a2 = assign[v2];
                float l0 = (s0 == 1) ? a0 : 1.0f - a0;
                float l1 = (s1 == 1) ? a1 : 1.0f - a1;
                float l2 = (s2 == 1) ? a2 : 1.0f - a2;
                float sat = fmaxf(fmaxf(l0, l1), l2);
                sat_out[c] = sat;
                localMin = fminf(localMin, sat);
                ind = (sat > 0.5f) ? (_Float16)1.0f : (_Float16)0.0f;
            }
            aFrag[t] = ind;
        }
    }

    // Exact satisfied-count for the whole wave in ONE v_wmma_f32_16x16x32_f16:
    // D = A(0/1 indicators, f16-exact) x B(ones) + 0, f32 accumulate.
    v16h ones;
#pragma unroll
    for (int t = 0; t < 16; ++t) ones[t] = (_Float16)1.0f;
    v8f acc = {};
    acc = __builtin_amdgcn_wmma_f32_16x16x32_f16(
        /*neg_a=*/false, aFrag, /*neg_b=*/false, ones,
        /*c_mod=*/(short)0, acc, /*reuse_a=*/false, /*reuse_b=*/false);

    // Every column of D equals the row-sum vector; a lane's 8 accumulators are one
    // half-column, its xor-16 partner holds the other half -> one swizzle finishes it.
    float s = acc[0] + acc[1] + acc[2] + acc[3] + acc[4] + acc[5] + acc[6] + acc[7];
    s += SWZ_F32(s, 0x401F); // xor 16

    // Wave min via ds_swizzle xor tree (wave32, all lanes active).
    float m = localMin;
    m = fminf(m, SWZ_F32(m, 0x401F)); // xor 16
    m = fminf(m, SWZ_F32(m, 0x201F)); // xor 8
    m = fminf(m, SWZ_F32(m, 0x101F)); // xor 4
    m = fminf(m, SWZ_F32(m, 0x081F)); // xor 2
    m = fminf(m, SWZ_F32(m, 0x041F)); // xor 1

    if ((tid & 31) == 0) {
        atomicMin((unsigned int*)min_slot, __float_as_uint(m)); // positive floats
        atomicAdd(cnt_slot, s);                                  // exact integer-valued f32, order-independent
    }
}

extern "C" void kernel_launch(void* const* d_in, const int* in_sizes, int n_in,
                              void* d_out, int out_size, void* d_ws, size_t ws_size,
                              hipStream_t stream) {
    const float* logits = (const float*)d_in[0];
    const int*   vic    = (const int*)d_in[1];
    const int*   signs  = (const int*)d_in[2];
    float* out = (float*)d_out;

    const int n_vars    = in_sizes[0];
    const int n_clauses = in_sizes[1] / 3;

    float* assignments = out;                       // [n_vars]
    float* clause_sat  = out + n_vars;              // [n_clauses]
    float* min_slot    = out + n_vars + n_clauses;  // all_satisfied
    float* cnt_slot    = min_slot + 1;              // n_satisfied

    const int sgBlocks = (n_vars + 255) / 256;
    sat_sigmoid_init<<<sgBlocks, 256, 0, stream>>>(
        logits, out, n_vars, (long long)n_vars + n_clauses);

    const int clausesPerBlock = BLOCK * CPT;
    const int clBlocks = (n_clauses + clausesPerBlock - 1) / clausesPerBlock;
    sat_clause_kernel<<<clBlocks, BLOCK, 0, stream>>>(
        assignments, vic, signs, clause_sat, min_slot, cnt_slot, n_clauses);
}
